// LSTMDecoder_85770496901255
// MI455X (gfx1250) — matrix operand order repaired and checked by reading, not compiled, not run on hardware
//
#include <hip/hip_runtime.h>
#include <hip/hip_bf16.h>

// ---------------------------------------------------------------------------
// LSTM decoder w/ location attention for MI455X (gfx1250, wave32, WMMA).
// - all GEMMs through v_wmma_f32_16x16x32_bf16 (f32 accumulate), weights
//   converted to bf16 once per launch and kept L2-resident (~45MB << 192MB L2)
// - A-tile staged into LDS via the Tensor Data Mover (tensor_load_to_lds +
//   s_wait_tensorcnt) when the builtin is available; cooperative b128 loads
//   otherwise
// - serial per-step scalar work (conv/softmax/LN/gates) in small VALU kernels
// ---------------------------------------------------------------------------

typedef unsigned short u16;
typedef unsigned int   u32;
typedef unsigned long long u64;
typedef __attribute__((ext_vector_type(16))) __bf16 bf16x16;
typedef __attribute__((ext_vector_type(8)))  float  f32x8;
typedef __attribute__((ext_vector_type(4))) unsigned int u32x4;
typedef __attribute__((ext_vector_type(8))) int i32x8;
typedef __attribute__((ext_vector_type(4))) int i32x4;

#define BB 32
#define TT 512
#define LL 128
#define DD 1024
#define EE 1024
#define VV 10240
#define AA 512
#define CC 10
#define KK 101

__device__ __forceinline__ u16 f2bf(float f) {
  u32 u = __float_as_uint(f);
  u += 0x7FFFu + ((u >> 16) & 1u);   // round-to-nearest-even
  return (u16)(u >> 16);
}
__device__ __forceinline__ float bf2f(u16 h) {
  return __uint_as_float(((u32)h) << 16);
}

union FragAB { bf16x16 v; uint4 q[2]; };
union FragC  { f32x8  v; float f[8]; };

// ---------------------------------------------------------------------------
// Generic skinny GEMM: OUT(M,N) = A(M,1024) * W(N,1024)^T  (+bias)(+accum)(tanh)
// A,W bf16 row-major, K fixed = 1024. Block = 128 threads (4 waves), each wave
// owns 32 columns (2 WMMA n-tiles), block owns 128 columns and one 32-row
// m-tile staged in LDS. Fragment layouts per CDNA5 ISA 7.12.2.
// flags: bit0 = accumulate from OUT, bit1 = tanh epilogue.
// ---------------------------------------------------------------------------
__global__ __launch_bounds__(128)
void k_gemm_bf16(const u16* __restrict__ Abf, const u16* __restrict__ Wbf,
                 const float* __restrict__ bias,
                 float* __restrict__ OUT, int ldout,
                 u16* __restrict__ OUT16, int ldo16, int flags)
{
  __shared__ u16 sA[32 * 1024];                 // 64KB A tile
  const int tid  = threadIdx.x;
  const int wv   = tid >> 5;
  const int lane = tid & 31;
  const int lg   = lane >> 4;                   // lane group (0: lanes 0-15)
  const int ll   = lane & 15;
  const int nBase = blockIdx.x * 128 + wv * 32;
  const size_t row0 = (size_t)blockIdx.y * 32;

#if __has_builtin(__builtin_amdgcn_tensor_load_to_lds)
  // ---- TDM stage: one contiguous 64KB tile, 1-D descriptor, 8B elements ----
  if (wv == 0) {
    const u64 ga = (u64)(const void*)(Abf + row0 * 1024);
    const u32 lds_base = (u32)(size_t)(&sA[0]);
    // D# group0: count=1 | lds_addr | global_addr | type=2
    u32x4 g0;
    g0[0] = 1u;                                    // count=1 (valid descriptor)
    g0[1] = lds_base;                              // lds_addr [63:32]
    g0[2] = (u32)ga;                               // global_addr [95:64]
    g0[3] = (u32)((ga >> 32) & 0x1FFFFFFu) | (2u << 30);  // addr msbs | type=2
    // D# group1: data_size=3 (8B); tensor_dim0=8192, dim1=1; tile_dim0=8192,
    // tile_dim1=1; tensor_dim0_stride=8192, tensor_dim1_stride=8192
    i32x8 g1;
    g1[0] = (int)(3u << 16);                       // wg_mask=0 | data_size=3
    g1[1] = (int)((8192u & 0xFFFFu) << 16);        // tensor_dim0[15:0]
    g1[2] = (int)((8192u >> 16) | (1u << 16));     // tensor_dim0[31:16]|dim1 lo
    g1[3] = (int)(8192u << 16);                    // dim1 hi | tile_dim0
    g1[4] = 1;                                     // tile_dim1=1 | tile_dim2=0
    g1[5] = 8192;                                  // tensor_dim0_stride[31:0]
    g1[6] = (int)((8192u & 0xFFFFu) << 16);        // str0 hi | str1[15:0]
    g1[7] = 0;                                     // tensor_dim1_stride[47:16]
    i32x4 g2 = {0, 0, 0, 0};
    i32x4 g3 = {0, 0, 0, 0};
#if defined(__clang_major__) && (__clang_major__ >= 23)
    i32x8 g4 = {0, 0, 0, 0, 0, 0, 0, 0};
    __builtin_amdgcn_tensor_load_to_lds(g0, g1, g2, g3, g4, 0);
#else
    __builtin_amdgcn_tensor_load_to_lds(g0, g1, g2, g3, 0);
#endif
    __builtin_amdgcn_s_wait_tensorcnt(0);
  }
  __syncthreads();
#else
  { // cooperative stage of A tile: 32 rows x 1024 bf16 = 4096 uint4
    const uint4* gA  = reinterpret_cast<const uint4*>(Abf + row0 * 1024);
    uint4*       sA4 = reinterpret_cast<uint4*>(sA);
    for (int i = tid; i < 4096; i += 128) sA4[i] = gA[i];
  }
  __syncthreads();
#endif

  FragC acc[2][2];
  #pragma unroll
  for (int mt = 0; mt < 2; ++mt)
    #pragma unroll
    for (int nt = 0; nt < 2; ++nt)
      #pragma unroll
      for (int j = 0; j < 8; ++j) {
        const int m = mt * 16 + lg * 8 + j;       // C layout: VGPR j -> M
        const int n = nBase + nt * 16 + ll;       // N striped over lanes
        float c0 = 0.f;
        if (flags & 1) c0 = OUT[(row0 + m) * (size_t)ldout + n];
        if (bias)      c0 += bias[n];
        acc[mt][nt].f[j] = c0;
      }

  for (int k0 = 0; k0 < 1024; k0 += 32) {
    FragAB a[2], b[2];
    #pragma unroll
    for (int mt = 0; mt < 2; ++mt) {
      // A 16x32 bf16 layout: lane<16 -> row=ll, K in {k0..k0+7, k0+16..k0+23}
      //                      lane>=16 -> K halves shifted by 8
      const u16* sp = &sA[(size_t)(mt * 16 + ll) * 1024 + k0 + lg * 8];
      a[mt].q[0] = *reinterpret_cast<const uint4*>(sp);
      a[mt].q[1] = *reinterpret_cast<const uint4*>(sp + 16);
    }
    #pragma unroll
    for (int nt = 0; nt < 2; ++nt) {
      // B 32x16 bf16 layout: lane owns column n, K same as A: {k0+8*lg..+7, k0+16+8*lg..+7}
      const u16* wp = Wbf + (size_t)(nBase + nt * 16 + ll) * 1024 + k0 + lg * 8;
      b[nt].q[0] = *reinterpret_cast<const uint4*>(wp);
      b[nt].q[1] = *reinterpret_cast<const uint4*>(wp + 16);
      __builtin_prefetch(wp + 128, 0, 1);        // global_prefetch_b8 ahead
    }
    #pragma unroll
    for (int mt = 0; mt < 2; ++mt)
      #pragma unroll
      for (int nt = 0; nt < 2; ++nt)
        acc[mt][nt].v = __builtin_amdgcn_wmma_f32_16x16x32_bf16(
            false, a[mt].v, false, b[nt].v, (short)0, acc[mt][nt].v,
            false, false);
  }

  #pragma unroll
  for (int mt = 0; mt < 2; ++mt)
    #pragma unroll
    for (int nt = 0; nt < 2; ++nt)
      #pragma unroll
      for (int j = 0; j < 8; ++j) {
        const int m = mt * 16 + lg * 8 + j;
        const int n = nBase + nt * 16 + ll;
        float v = acc[mt][nt].f[j];
        if (flags & 2) v = tanhf(v);
        if (OUT)   OUT[(row0 + m) * (size_t)ldout + n] = v;
        if (OUT16) OUT16[(row0 + m) * (size_t)ldo16 + n] = f2bf(v);
      }
}

// --------------------------- small VALU kernels ----------------------------

__global__ void k_f32_to_bf16(const float* __restrict__ in, u16* __restrict__ out, int n) {
  int i = blockIdx.x * 256 + threadIdx.x;
  if (i < n) out[i] = f2bf(in[i]);
}

__global__ void k_zero(float* __restrict__ p, int n) {
  int i = blockIdx.x * 256 + threadIdx.x;
  if (i < n) p[i] = 0.f;
}

// transpose att_U (A,C) -> Ut (C,A) so k_att lanes read coalesced
__global__ void k_transpose_U(const float* __restrict__ U, float* __restrict__ Ut) {
  int a = blockIdx.x * 256 + threadIdx.x;
  if (a < AA)
    for (int c = 0; c < CC; ++c) Ut[c * AA + a] = U[a * CC + c];
}

// conv1d(1->C, K=101, pad 50) over alpha(B,T); one block = 256 t's of one b.
__global__ __launch_bounds__(256)
void k_conv(const float* __restrict__ alpha, const float* __restrict__ cw,
            float* __restrict__ f)
{
  __shared__ float sAl[256 + KK - 1];
  const int b = blockIdx.x >> 1;
  const int t0 = (blockIdx.x & 1) * 256;
  const int tid = threadIdx.x;
  for (int i = tid; i < 256 + KK - 1; i += 256) {
    int t = t0 + i - KK / 2;
    sAl[i] = (t >= 0 && t < TT) ? alpha[b * TT + t] : 0.f;
  }
  __syncthreads();
  const int t = t0 + tid;
  for (int c = 0; c < CC; ++c) {
    float acc = 0.f;
    #pragma unroll
    for (int k = 0; k < KK; ++k) acc += cw[c * KK + k] * sAl[tid + k];
    f[(b * CC + c) * TT + t] = acc;
  }
}

// e[b,t] = sum_a tanh(Vh[b,t,a] + sW[b,a] + f[b,:,t].Ut[:,a]) * w[a]
// one wave per (b,t); lanes stride A; shuffle-reduce.
__global__ __launch_bounds__(256)
void k_att(const u16* __restrict__ Vh16, const float* __restrict__ sW,
           const float* __restrict__ f, const float* __restrict__ Ut,
           const float* __restrict__ w, float* __restrict__ e)
{
  const int wid  = blockIdx.x * 8 + (threadIdx.x >> 5);
  const int lane = threadIdx.x & 31;
  const int b = wid / TT, t = wid % TT;
  float fc[CC];
  #pragma unroll
  for (int c = 0; c < CC; ++c) fc[c] = f[(b * CC + c) * TT + t];
  const u16*   vh = Vh16 + (size_t)(b * TT + t) * AA;
  const float* sw = sW + b * AA;
  float part = 0.f;
  for (int a = lane; a < AA; a += 32) {
    float u = 0.f;
    #pragma unroll
    for (int c = 0; c < CC; ++c) u += fc[c] * Ut[c * AA + a];
    part += tanhf(bf2f(vh[a]) + sw[a] + u) * w[a];
  }
  #pragma unroll
  for (int off = 16; off > 0; off >>= 1) part += __shfl_xor(part, off, 32);
  if (lane == 0) e[b * TT + t] = part;
}

__global__ __launch_bounds__(256)
void k_softmax(const float* __restrict__ e, float* __restrict__ alpha)
{
  __shared__ float red[256];
  const int b = blockIdx.x, tid = threadIdx.x;
  const float v0 = e[b * TT + tid];
  const float v1 = e[b * TT + tid + 256];
  red[tid] = fmaxf(v0, v1); __syncthreads();
  for (int s = 128; s > 0; s >>= 1) { if (tid < s) red[tid] = fmaxf(red[tid], red[tid + s]); __syncthreads(); }
  const float mx = red[0]; __syncthreads();
  const float e0 = __expf(v0 - mx), e1 = __expf(v1 - mx);
  red[tid] = e0 + e1; __syncthreads();
  for (int s = 128; s > 0; s >>= 1) { if (tid < s) red[tid] += red[tid + s]; __syncthreads(); }
  const float inv = 1.f / red[0];
  alpha[b * TT + tid]       = e0 * inv;
  alpha[b * TT + tid + 256] = e1 * inv;
}

// g[b,e] = sum_t alpha[b,t] * hb[b,t,e]; block per b, thread owns 4 e's.
__global__ __launch_bounds__(256)
void k_context(const float* __restrict__ alpha, const u16* __restrict__ hb,
               float* __restrict__ g, u16* __restrict__ g16)
{
  const int b = blockIdx.x, tid = threadIdx.x;
  float acc[4] = {0.f, 0.f, 0.f, 0.f};
  const u16* hbb = hb + (size_t)b * TT * EE;
  for (int t = 0; t < TT; ++t) {
    const float a = alpha[b * TT + t];
    #pragma unroll
    for (int q = 0; q < 4; ++q)
      acc[q] += a * bf2f(hbb[(size_t)t * EE + tid + q * 256]);
  }
  #pragma unroll
  for (int q = 0; q < 4; ++q) {
    g[b * EE + tid + q * 256]   = acc[q];
    g16[b * EE + tid + q * 256] = f2bf(acc[q]);
  }
}

// LayerNorm x3 over 4096 + LSTM gate update; block per b.
__global__ __launch_bounds__(256)
void k_lstm(const int* __restrict__ targets, int t,
            const float* __restrict__ emb_ys,
            const float* __restrict__ ws0, const float* __restrict__ ws1,
            const float* __restrict__ ln0g, const float* __restrict__ ln0b,
            const float* __restrict__ ln1g, const float* __restrict__ ln1b,
            const float* __restrict__ ln2g, const float* __restrict__ ln2b,
            float* __restrict__ s, float* __restrict__ c, u16* __restrict__ s16)
{
  __shared__ float red[256];
  const int b = blockIdx.x, tid = threadIdx.x;
  const float* emb = emb_ys + (size_t)targets[b * LL + t] * 4096;
  const float* w0  = ws0 + (size_t)b * 4096;
  const float* w1  = ws1 + (size_t)b * 4096;
  float sm[3] = {0, 0, 0}, sq[3] = {0, 0, 0};
  for (int j = tid; j < 4096; j += 256) {
    float a0 = emb[j], a1 = w0[j], a2 = w1[j];
    sm[0] += a0; sq[0] += a0 * a0;
    sm[1] += a1; sq[1] += a1 * a1;
    sm[2] += a2; sq[2] += a2 * a2;
  }
  float mu[3], rs[3];
  for (int v = 0; v < 3; ++v) {
    red[tid] = sm[v]; __syncthreads();
    for (int st = 128; st > 0; st >>= 1) { if (tid < st) red[tid] += red[tid + st]; __syncthreads(); }
    const float tot = red[0]; __syncthreads();
    red[tid] = sq[v]; __syncthreads();
    for (int st = 128; st > 0; st >>= 1) { if (tid < st) red[tid] += red[tid + st]; __syncthreads(); }
    const float tsq = red[0]; __syncthreads();
    mu[v] = tot / 4096.f;
    rs[v] = rsqrtf(tsq / 4096.f - mu[v] * mu[v] + 1e-5f);
  }
  for (int d = tid; d < 1024; d += 256) {
    float rec[4];
    #pragma unroll
    for (int q = 0; q < 4; ++q) {
      const int j = q * 1024 + d;
      rec[q] = (emb[j] - mu[0]) * rs[0] * ln0g[j] + ln0b[j]
             + (w0[j]  - mu[1]) * rs[1] * ln1g[j] + ln1b[j]
             + (w1[j]  - mu[2]) * rs[2] * ln2g[j] + ln2b[j];
    }
    const float ig = 1.f / (1.f + __expf(-rec[0]));
    const float fg = 1.f / (1.f + __expf(-rec[1]));
    const float cg = tanhf(rec[2]);
    const float og = 1.f / (1.f + __expf(-rec[3]));
    const float cn = fg * c[b * 1024 + d] + ig * cg;
    const float sn = og * tanhf(cn);
    c[b * 1024 + d] = cn;
    s[b * 1024 + d] = sn;
    s16[b * 1024 + d] = f2bf(sn);
  }
}

// ---------------------------------------------------------------------------

extern "C" void kernel_launch(void* const* d_in, const int* in_sizes, int n_in,
                              void* d_out, int out_size, void* d_ws, size_t ws_size,
                              hipStream_t stream)
{
  (void)in_sizes; (void)n_in; (void)out_size; (void)ws_size;
  const int*   targets  = (const int*)  d_in[0];
  const float* hbatch   = (const float*)d_in[1];
  const float* emb_ys   = (const float*)d_in[4];
  const float* W_sy     = (const float*)d_in[5];
  const float* W_gy     = (const float*)d_in[6];
  const float* b_gy     = (const float*)d_in[7];
  const float* W_yy     = (const float*)d_in[8];
  const float* b_yy     = (const float*)d_in[9];
  const float* W_ss     = (const float*)d_in[10];
  const float* W_gs     = (const float*)d_in[11];
  const float* b_gs     = (const float*)d_in[12];
  const float* ln0g     = (const float*)d_in[13];
  const float* ln0b     = (const float*)d_in[14];
  const float* ln1g     = (const float*)d_in[15];
  const float* ln1b     = (const float*)d_in[16];
  const float* ln2g     = (const float*)d_in[17];
  const float* ln2b     = (const float*)d_in[18];
  const float* att_W    = (const float*)d_in[19];
  const float* att_V    = (const float*)d_in[20];
  const float* att_b    = (const float*)d_in[21];
  const float* att_U    = (const float*)d_in[22];
  const float* att_conv = (const float*)d_in[23];
  const float* att_w    = (const float*)d_in[24];
  float* out = (float*)d_out;

  char* p = (char*)d_ws;
  auto alloc = [&](size_t bytes) -> char* {
    char* r = p; p += (bytes + 255) & ~(size_t)255; return r;
  };
  u16*   hb16   = (u16*)alloc((size_t)BB * TT * EE * 2);
  u16*   Wyy16  = (u16*)alloc((size_t)VV * DD * 2);
  u16*   Wss16  = (u16*)alloc((size_t)4 * DD * DD * 2);
  u16*   Wgs16  = (u16*)alloc((size_t)4 * DD * EE * 2);
  u16*   Wsy16  = (u16*)alloc((size_t)DD * DD * 2);
  u16*   Wgy16  = (u16*)alloc((size_t)DD * EE * 2);
  u16*   attW16 = (u16*)alloc((size_t)AA * DD * 2);
  u16*   attV16 = (u16*)alloc((size_t)AA * EE * 2);
  u16*   Vh16   = (u16*)alloc((size_t)BB * TT * AA * 2);
  float* Ut     = (float*)alloc((size_t)CC * AA * 4);
  float* sWbuf  = (float*)alloc((size_t)BB * AA * 4);
  float* fbuf   = (float*)alloc((size_t)BB * CC * TT * 4);
  float* ebuf   = (float*)alloc((size_t)BB * TT * 4);
  float* alphab = (float*)alloc((size_t)BB * TT * 4);
  float* gbuf   = (float*)alloc((size_t)BB * EE * 4);
  u16*   g16    = (u16*)alloc((size_t)BB * EE * 2);
  float* sbuf   = (float*)alloc((size_t)BB * DD * 4);
  float* cbuf   = (float*)alloc((size_t)BB * DD * 4);
  u16*   s16    = (u16*)alloc((size_t)BB * DD * 2);
  float* tmpz   = (float*)alloc((size_t)BB * DD * 4);
  u16*   z16    = (u16*)alloc((size_t)BB * DD * 2);
  float* ws0    = (float*)alloc((size_t)BB * 4 * DD * 4);
  float* ws1    = (float*)alloc((size_t)BB * 4 * DD * 4);

  auto cvt = [&](const float* src, u16* dst, size_t n) {
    k_f32_to_bf16<<<dim3((unsigned)((n + 255) / 256)), 256, 0, stream>>>(src, dst, (int)n);
  };
  // one-time (per launch) conversions — weights then stay L2-resident in bf16
  cvt(hbatch, hb16,  (size_t)BB * TT * EE);
  cvt(W_yy,   Wyy16, (size_t)VV * DD);
  cvt(W_ss,   Wss16, (size_t)4 * DD * DD);
  cvt(W_gs,   Wgs16, (size_t)4 * DD * EE);
  cvt(W_sy,   Wsy16, (size_t)DD * DD);
  cvt(W_gy,   Wgy16, (size_t)DD * EE);
  cvt(att_W,  attW16,(size_t)AA * DD);
  cvt(att_V,  attV16,(size_t)AA * EE);
  k_transpose_U<<<dim3((AA + 255) / 256), 256, 0, stream>>>(att_U, Ut);

  // Vh = hbatch @ att_V.T + att_b  -> bf16 (B*T, A)
  k_gemm_bf16<<<dim3(AA / 128, (BB * TT) / 32), 128, 0, stream>>>(
      hb16, attV16, att_b, nullptr, 0, Vh16, AA, 0);

  // zero initial state s, c, alpha (and s16)
  k_zero<<<dim3((BB * DD + 255) / 256), 256, 0, stream>>>(sbuf, BB * DD);
  k_zero<<<dim3((BB * DD + 255) / 256), 256, 0, stream>>>(cbuf, BB * DD);
  k_zero<<<dim3((BB * TT + 255) / 256), 256, 0, stream>>>(alphab, BB * TT);
  k_zero<<<dim3((BB * DD / 2 + 255) / 256), 256, 0, stream>>>((float*)s16, BB * DD / 2);

  for (int t = 0; t < LL; ++t) {
    // location attention
    k_conv<<<dim3(BB * (TT / 256)), 256, 0, stream>>>(alphab, att_conv, fbuf);
    k_gemm_bf16<<<dim3(AA / 128, 1), 128, 0, stream>>>(
        s16, attW16, nullptr, sWbuf, AA, nullptr, 0, 0);
    k_att<<<dim3(BB * TT / 8), 256, 0, stream>>>(Vh16, sWbuf, fbuf, Ut, att_w, ebuf);
    k_softmax<<<dim3(BB), 256, 0, stream>>>(ebuf, alphab);
    k_context<<<dim3(BB), 256, 0, stream>>>(alphab, hb16, gbuf, g16);
    // output head: z = tanh(s@W_sy.T + g@W_gy.T + b_gy); y = z@W_yy.T + b_yy
    k_gemm_bf16<<<dim3(DD / 128, 1), 128, 0, stream>>>(
        s16, Wsy16, nullptr, tmpz, DD, nullptr, 0, 0);
    k_gemm_bf16<<<dim3(DD / 128, 1), 128, 0, stream>>>(
        g16, Wgy16, b_gy, tmpz, DD, z16, DD, 1 | 2);
    k_gemm_bf16<<<dim3(VV / 128, 1), 128, 0, stream>>>(
        z16, Wyy16, b_yy, out + (size_t)t * VV, LL * VV, nullptr, 0, 0);
    // gate pre-activations
    k_gemm_bf16<<<dim3(4 * DD / 128, 1), 128, 0, stream>>>(
        s16, Wss16, nullptr, ws0, 4 * DD, nullptr, 0, 0);
    k_gemm_bf16<<<dim3(4 * DD / 128, 1), 128, 0, stream>>>(
        g16, Wgs16, b_gs, ws1, 4 * DD, nullptr, 0, 0);
    // LN + LSTM update (writes s, c, s16 for next step)
    k_lstm<<<dim3(BB), 256, 0, stream>>>(targets, t, emb_ys, ws0, ws1,
        ln0g, ln0b, ln1g, ln1b, ln2g, ln2b, sbuf, cbuf, s16);
  }
}
